// Mamba_52974126629351
// MI455X (gfx1250) — matrix operand order repaired and checked by reading, not compiled, not run on hardware
//
#include <hip/hip_runtime.h>
#include <hip/hip_bf16.h>

typedef __attribute__((ext_vector_type(16))) __bf16 v16bf;
typedef __attribute__((ext_vector_type(8)))  __bf16 v8bf;
typedef __attribute__((ext_vector_type(4)))  __bf16 bf16x4;
typedef __attribute__((ext_vector_type(8)))  float  v8f;
typedef __attribute__((ext_vector_type(4)))  float  f32x4;
typedef __attribute__((ext_vector_type(2)))  float  v2f;
typedef __attribute__((ext_vector_type(4)))  int    v4i;

#define B_SZ 2
#define L_SZ 2048
#define H_SZ 2048
#define I_SZ 4096
#define NS   16
#define R_SZ 128
#define M_SZ (B_SZ * L_SZ)          /* 4096 rows (b,l) */
#define TWO_I (2 * I_SZ)            /* 8192 */
#define SPW  (R_SZ + 2 * NS)        /* 160 */
#define EPSF 1e-5f

#define GLOBAL_AS __attribute__((address_space(1)))
#define LDS_AS    __attribute__((address_space(3)))

#if defined(__HIP_DEVICE_COMPILE__) && __has_builtin(__builtin_amdgcn_global_load_async_to_lds_b128)
#define HAS_ASYNC_LDS 1
#else
#define HAS_ASYNC_LDS 0
#endif

// ---------------------------------------------------------------------------
// f32 -> bf16 bulk conversion (memory bound, vectorized 4-wide).
// ---------------------------------------------------------------------------
__global__ __launch_bounds__(256)
void cvt_f32_to_bf16(const float* __restrict__ src, __bf16* __restrict__ dst) {
    size_t idx = (size_t)blockIdx.x * 256 + threadIdx.x;
    f32x4 v = ((const f32x4*)src)[idx];
    bf16x4 o;
#pragma unroll
    for (int j = 0; j < 4; ++j) o[j] = (__bf16)v[j];
    ((bf16x4*)dst)[idx] = o;
}

// ---------------------------------------------------------------------------
// Big GEMM: C[M,N] = A[M,K](bf16) * W[N,K]^T(bf16), f32 accumulate.
// Block = 256 thr (8 waves, 2Mx4N), wave tile 32x32 (2x2 frags), block 64x128.
// W tile (128x32 bf16 = 8 KB) is double-buffered in LDS via async loads
// (ASYNCcnt path) when available; A frags are direct bf16 b128 global loads.
// Fragment layouts per CDNA5 ISA 7.12.2 (wave32):
//   A 16x32 bf16: m=lane&15, hi=lane>>4; halves 0..7 = K[8*hi..], 8..15 = K[16+8*hi..]
//   B 32x16 bf16: n=lane&15; halves 0..7 = K[8*hi..], 8..15 = K[16+8*hi..] of col n
//   C 16x16 f32 : vgpr v -> row v+8*hi, col = lane&15
// ---------------------------------------------------------------------------
__global__ __launch_bounds__(256)
void gemm_bf16_big(const __bf16* __restrict__ A, const __bf16* __restrict__ W,
                   float* __restrict__ C, int N, int K) {
    const int tid  = threadIdx.x;
    const int lane = tid & 31;
    const int wave = tid >> 5;
    const int wm = wave >> 2, wn = wave & 3;
    const int m = lane & 15, hi = lane >> 4;
    const int tM  = blockIdx.y * 64 + wm * 32;
    const int tNb = blockIdx.x * 128;
    const int tN  = tNb + wn * 32;

    v8f acc[2][2] = {};
    const __bf16* a0 = A + (size_t)(tM + m) * K;
    const __bf16* a1 = a0 + (size_t)16 * K;

#if HAS_ASYNC_LDS
    __shared__ __attribute__((aligned(16))) __bf16 wtile[2][128 * 32];
    auto stage = [&](int buf, int k0) {
#pragma unroll
        for (int r = 0; r < 2; ++r) {
            int c   = tid + 256 * r;          // 512 chunks of 8 halves
            int row = c >> 2;
            int ko  = (c & 3) * 8;
            const __bf16* gp = W + (size_t)(tNb + row) * K + k0 + ko;
            __bf16* lp = &wtile[buf][row * 32 + ko];
            __builtin_amdgcn_global_load_async_to_lds_b128(
                (GLOBAL_AS v4i*)gp, (LDS_AS v4i*)lp, 0, 0);
        }
    };
    stage(0, 0);
    for (int k0 = 0, it = 0; k0 < K; k0 += 32, ++it) {
        asm volatile("s_wait_asynccnt 0x0" ::: "memory");
        __syncthreads();
        if (k0 + 32 < K) stage((it + 1) & 1, k0 + 32);

        v16bf af[2], bf[2];
        __builtin_prefetch(a0 + k0 + 64, 0, 3);
#pragma unroll
        for (int f = 0; f < 2; ++f) {
            const __bf16* p = (f ? a1 : a0) + k0 + 8 * hi;
            *(v8bf*)&af[f]       = *(const v8bf*)p;
            *((v8bf*)&af[f] + 1) = *(const v8bf*)(p + 16);
        }
        const __bf16* wl = &wtile[it & 1][0];
#pragma unroll
        for (int in = 0; in < 2; ++in) {
            const __bf16* q = wl + (wn * 32 + in * 16 + m) * 32 + 8 * hi;
            *(v8bf*)&bf[in]       = *(const v8bf*)q;
            *((v8bf*)&bf[in] + 1) = *(const v8bf*)(q + 16);
        }
#pragma unroll
        for (int im = 0; im < 2; ++im)
#pragma unroll
            for (int in = 0; in < 2; ++in)
                acc[im][in] = __builtin_amdgcn_wmma_f32_16x16x32_bf16(
                    false, af[im], false, bf[in], (short)0, acc[im][in], false, false);
    }
#else
    const __bf16* w0 = W + (size_t)(tN + m) * K;
    const __bf16* w1 = w0 + (size_t)16 * K;
    for (int k0 = 0; k0 < K; k0 += 32) {
        v16bf af[2], bf[2];
        __builtin_prefetch(a0 + k0 + 64, 0, 3);
        __builtin_prefetch(w0 + k0 + 64, 0, 3);
#pragma unroll
        for (int f = 0; f < 2; ++f) {
            const __bf16* p = (f ? a1 : a0) + k0 + 8 * hi;
            *(v8bf*)&af[f]       = *(const v8bf*)p;
            *((v8bf*)&af[f] + 1) = *(const v8bf*)(p + 16);
            const __bf16* q = (f ? w1 : w0) + k0 + 8 * hi;
            *(v8bf*)&bf[f]       = *(const v8bf*)q;
            *((v8bf*)&bf[f] + 1) = *(const v8bf*)(q + 16);
        }
#pragma unroll
        for (int im = 0; im < 2; ++im)
#pragma unroll
            for (int in = 0; in < 2; ++in)
                acc[im][in] = __builtin_amdgcn_wmma_f32_16x16x32_bf16(
                    false, af[im], false, bf[in], (short)0, acc[im][in], false, false);
    }
#endif

#pragma unroll
    for (int im = 0; im < 2; ++im) {
        float* cp = C + (size_t)(tM + im * 16 + 8 * hi) * N + tN + m;
#pragma unroll
        for (int v = 0; v < 8; ++v) {
            cp[0]  = acc[im][0][v];
            cp[16] = acc[im][1][v];
            cp += N;
        }
    }
}

// ---------------------------------------------------------------------------
// Depthwise causal conv1d (K=4) + bias + SiLU.
// proj row-major [M_SZ, 2I]: h = cols [0,I). Output hconv [M_SZ, I].
// ---------------------------------------------------------------------------
__global__ __launch_bounds__(256)
void conv_silu_kernel(const float* __restrict__ proj, const float* __restrict__ cw,
                      const float* __restrict__ cb, float* __restrict__ hc) {
    int idx = blockIdx.x * 256 + threadIdx.x;        // [0, M_SZ*I)
    int i  = idx & (I_SZ - 1);
    int bl = idx >> 12;
    int l  = bl & (L_SZ - 1);
    f32x4 w = *(const f32x4*)(cw + i * 4);
    float acc = cb[i];
#pragma unroll
    for (int k = 0; k < 4; ++k) {
        int ls = l - 3 + k;
        if (ls >= 0)
            acc += w[k] * proj[(size_t)(bl - 3 + k) * TWO_I + i];
    }
    hc[idx] = acc / (1.f + __expf(-acc));            // silu
}

// ---------------------------------------------------------------------------
// fp32 WMMA 16x16x4 tile GEMM (x_proj): one wave per 16x16 C tile.
// ---------------------------------------------------------------------------
__global__ __launch_bounds__(32)
void gemm_f32_tile(const float* __restrict__ A, int lda,
                   const float* __restrict__ W, int ldw,
                   float* __restrict__ C, int ldc, int K) {
    const int lane = threadIdx.x;
    const int m = lane & 15, hi = lane >> 4;
    const int tM = blockIdx.y << 4, tN = blockIdx.x << 4;
    const float* ap = A + (size_t)(tM + m) * lda + 2 * hi;
    const float* wp = W + (size_t)(tN + m) * ldw + 2 * hi;
    v8f acc = {};
#pragma unroll 8
    for (int k0 = 0; k0 < K; k0 += 4) {
        v2f a = *(const v2f*)(ap + k0);
        v2f b = *(const v2f*)(wp + k0);
        acc = __builtin_amdgcn_wmma_f32_16x16x4_f32(false, a, false, b,
                                                    (short)0, acc, false, false);
    }
    float* cp = C + (size_t)(tM + 8 * hi) * ldc + tN + m;
#pragma unroll
    for (int v = 0; v < 8; ++v) { *cp = acc[v]; cp += ldc; }
}

// ---------------------------------------------------------------------------
// RMS norms on the [M_SZ, 160] x_proj rows.
// ---------------------------------------------------------------------------
__global__ __launch_bounds__(128)
void rmsnorm_kernel(const float* __restrict__ sp, const float* __restrict__ wdt,
                    const float* __restrict__ wb, const float* __restrict__ wc,
                    float* __restrict__ spn) {
    __shared__ float red[128];
    const int row = blockIdx.x, tid = threadIdx.x;
    const float* rp = sp + (size_t)row * SPW;
    float v = rp[tid];
    red[tid] = v * v;
    __syncthreads();
    for (int off = 64; off > 0; off >>= 1) {
        if (tid < off) red[tid] += red[tid + off];
        __syncthreads();
    }
    float inv = __frsqrt_rn(red[0] * (1.f / R_SZ) + EPSF);
    spn[(size_t)row * SPW + tid] = v * inv * wdt[tid];
    if (tid < 32) {
        int n = tid & 15, isC = tid >> 4;
        float x = rp[R_SZ + isC * NS + n];
        float ss = x * x;
        ss += __shfl_xor(ss, 8);
        ss += __shfl_xor(ss, 4);
        ss += __shfl_xor(ss, 2);
        ss += __shfl_xor(ss, 1);
        float invn = __frsqrt_rn(ss * (1.f / NS) + EPSF);
        const float* w = isC ? wc : wb;
        spn[(size_t)row * SPW + R_SZ + isC * NS + n] = x * invn * w[n];
    }
}

// ---------------------------------------------------------------------------
// dt GEMM (fp32 WMMA, K=R=128) + dt_bias + softplus -> delta.
// ---------------------------------------------------------------------------
__global__ __launch_bounds__(32)
void gemm_f32_softplus(const float* __restrict__ A, int lda,
                       const float* __restrict__ W, int ldw,
                       const float* __restrict__ bias,
                       float* __restrict__ C, int ldc, int K) {
    const int lane = threadIdx.x;
    const int m = lane & 15, hi = lane >> 4;
    const int tM = blockIdx.y << 4, tN = blockIdx.x << 4;
    const float* ap = A + (size_t)(tM + m) * lda + 2 * hi;
    const float* wp = W + (size_t)(tN + m) * ldw + 2 * hi;
    v8f acc = {};
#pragma unroll 8
    for (int k0 = 0; k0 < K; k0 += 4) {
        v2f a = *(const v2f*)(ap + k0);
        v2f b = *(const v2f*)(wp + k0);
        acc = __builtin_amdgcn_wmma_f32_16x16x4_f32(false, a, false, b,
                                                    (short)0, acc, false, false);
    }
    float bcol = bias[tN + m];
    float* cp = C + (size_t)(tM + 8 * hi) * ldc + tN + m;
#pragma unroll
    for (int v = 0; v < 8; ++v) {
        float x = acc[v] + bcol;
        *cp = (x > 20.f) ? x : log1pf(__expf(x));
        cp += ldc;
    }
}

// ---------------------------------------------------------------------------
// Selective scan: one thread per (b,i); N=16 state in registers.
// Fused: y = (scan + D*u) * silu(gate), stored as bf16 for the out-proj GEMM.
// ---------------------------------------------------------------------------
__global__ __launch_bounds__(256)
void scan_kernel(const float* __restrict__ u, const float* __restrict__ delta,
                 const float* __restrict__ spn, const float* __restrict__ Ap,
                 const float* __restrict__ Dp, const float* __restrict__ proj,
                 __bf16* __restrict__ y) {
    int idx = blockIdx.x * 256 + threadIdx.x;        // [0, B*I)
    int b = idx >> 12;
    int i = idx & (I_SZ - 1);
    float a[NS], s[NS];
#pragma unroll
    for (int n = 0; n < NS; ++n) { a[n] = Ap[i * NS + n]; s[n] = 0.f; }
    float Di = Dp[i];
    size_t rowu = (size_t)b * L_SZ * I_SZ + i;
    size_t rowp = (size_t)b * L_SZ * TWO_I + I_SZ + i;
    const float* bc = spn + (size_t)b * L_SZ * SPW + R_SZ;
    for (int l = 0; l < L_SZ; ++l) {
        float ut = u[rowu];
        float dt = delta[rowu];
        float yv = 0.f;
#pragma unroll
        for (int n = 0; n < NS; ++n) {
            float dA = __expf(dt * a[n]);
            s[n] = dA * s[n] + (dt * bc[n]) * ut;
            yv += s[n] * bc[NS + n];
        }
        float g = proj[rowp];
        y[rowu] = (__bf16)((yv + Di * ut) * (g / (1.f + __expf(-g))));
        rowu += I_SZ; rowp += TWO_I; bc += SPW;
    }
}

// ---------------------------------------------------------------------------
extern "C" void kernel_launch(void* const* d_in, const int* in_sizes, int n_in,
                              void* d_out, int out_size, void* d_ws, size_t ws_size,
                              hipStream_t stream) {
    const float* x       = (const float*)d_in[0];
    const float* W_in    = (const float*)d_in[1];
    const float* conv_w  = (const float*)d_in[2];
    const float* conv_b  = (const float*)d_in[3];
    const float* Wx      = (const float*)d_in[4];
    const float* dt_ln_w = (const float*)d_in[5];
    const float* b_ln_w  = (const float*)d_in[6];
    const float* c_ln_w  = (const float*)d_in[7];
    const float* Wdt     = (const float*)d_in[8];
    const float* dt_bias = (const float*)d_in[9];
    const float* A       = (const float*)d_in[10];
    const float* D       = (const float*)d_in[11];
    const float* Wout    = (const float*)d_in[12];
    float* out = (float*)d_out;

    float* ws    = (float*)d_ws;
    float* proj  = ws;                                   // [M, 2I] f32
    float* hconv = proj  + (size_t)M_SZ * TWO_I;         // [M, I]  f32
    float* sp    = hconv + (size_t)M_SZ * I_SZ;          // [M, 160]
    float* spn   = sp    + (size_t)M_SZ * SPW;           // [M, 160]
    float* delta = spn   + (size_t)M_SZ * SPW;           // [M, I]  f32
    __bf16* xbf    = (__bf16*)(delta + (size_t)M_SZ * I_SZ); // [M, H]
    __bf16* winbf  = xbf    + (size_t)M_SZ * H_SZ;           // [2I, H]
    __bf16* woutbf = winbf  + (size_t)TWO_I * H_SZ;          // [H, I]
    __bf16* ybf    = woutbf + (size_t)H_SZ * I_SZ;           // [M, I]

    // 0) one-shot f32 -> bf16 conversions for GEMM operands
    cvt_f32_to_bf16<<<(M_SZ * H_SZ) / 1024, 256, 0, stream>>>(x, xbf);
    cvt_f32_to_bf16<<<(TWO_I * H_SZ) / 1024, 256, 0, stream>>>(W_in, winbf);
    cvt_f32_to_bf16<<<(H_SZ * I_SZ) / 1024, 256, 0, stream>>>(Wout, woutbf);

    // 1) in_proj: proj[M, 2I] = x[M,H] * W_in[2I,H]^T  (bf16 WMMA, async-LDS W tiles)
    gemm_bf16_big<<<dim3(TWO_I / 128, M_SZ / 64), 256, 0, stream>>>(
        xbf, winbf, proj, TWO_I, H_SZ);

    // 2) depthwise causal conv + silu -> hconv[M, I]
    conv_silu_kernel<<<(M_SZ * I_SZ) / 256, 256, 0, stream>>>(
        proj, conv_w, conv_b, hconv);

    // 3) x_proj: sp[M,160] = hconv[M,I] * Wx[160,I]^T  (fp32 WMMA)
    gemm_f32_tile<<<dim3(SPW / 16, M_SZ / 16), 32, 0, stream>>>(
        hconv, I_SZ, Wx, I_SZ, sp, SPW, I_SZ);

    // 4) RMS norms -> spn[M,160]
    rmsnorm_kernel<<<M_SZ, 128, 0, stream>>>(sp, dt_ln_w, b_ln_w, c_ln_w, spn);

    // 5) delta[M,I] = softplus(spn[:, :128] * Wdt[I,128]^T + dt_bias)  (fp32 WMMA)
    gemm_f32_softplus<<<dim3(I_SZ / 16, M_SZ / 16), 32, 0, stream>>>(
        spn, SPW, Wdt, R_SZ, dt_bias, delta, I_SZ, R_SZ);

    // 6) selective scan + D*u + silu(gate) -> ybf[M, I] (bf16)
    scan_kernel<<<(B_SZ * I_SZ) / 256, 256, 0, stream>>>(
        hconv, delta, spn, A, D, proj, ybf);

    // 7) out_proj: out[M,H] = y[M,I] * Wout[H,I]^T  (bf16 WMMA, async-LDS W tiles)
    gemm_bf16_big<<<dim3(H_SZ / 128, M_SZ / 64), 256, 0, stream>>>(
        ybf, woutbf, out, H_SZ, I_SZ);
}
